// SphereIBSDE_6786048327948
// MI455X (gfx1250) — matrix operand order repaired and checked by reading, not compiled, not run on hardware
//
#include <hip/hip_runtime.h>
#include <hip/hip_bf16.h>

typedef __attribute__((ext_vector_type(16))) _Float16 v16h;
typedef __attribute__((ext_vector_type(8)))  _Float16 v8h;
typedef __attribute__((ext_vector_type(8)))  float    v8f;

#define NSTEP   100
#define BTOT    4096
#define HID     256
#define D3DIM   96
#define DXP     32
#define MT      16              // batch rows per block

// LDS strides padded to break bank conflicts
#define XSTR    100             // f32 xt/grad tiles: 100 mod 64 = 36 -> conflict-free
#define XHSTR   104             // f16 xt mirror: 208B rows, 52 dwords, distinct banks
#define HSTRH   264             // f16 hidden tiles: 528B = 132 dwords, 132 mod 64 = 4

// weight-fragment regions in d_ws (units: fragments of 512 halfs = 1KB)
#define FR_PER_STEP 224         // L1:48 [0,48) L2:128 [48,176) L3:48 [176,224)
#define FR_G_TOTAL  (NSTEP * FR_PER_STEP)   // 22400
#define FR_PL1      FR_G_TOTAL              // 22400..22448
#define FR_PL2      (FR_G_TOTAL + 48)       // 22448..22576
#define FR_TOTAL    (FR_G_TOTAL + 48 + 128) // 22576

// ---------------------------------------------------------------------------
// One-time weight packer: f32 [K][N] -> f16 WMMA B-fragments (32x16 per frag).
// B layout (matches A): lane l -> N = l%16; VGPR v holds K = 16*(v/4) + 8*(l/16)
// + 2*(v%4), +1.  Thread u writes dword u -> coalesced stores.
// ---------------------------------------------------------------------------
__global__ void pack_weights(const float* __restrict__ gW1,
                             const float* __restrict__ gW2,
                             const float* __restrict__ gW3,
                             const float* __restrict__ pW1,
                             const float* __restrict__ pW2,
                             _Float16* __restrict__ ws) {
  int b = blockIdx.x;
  const float* W; int nt, kb, fo;
  if (b < FR_G_TOTAL) {
    int s = b / FR_PER_STEP, r = b % FR_PER_STEP;
    if (r < 48)       { W = gW1 + (size_t)s * 96 * 256;  nt = r / 3;       kb = r % 3;       fo = 256; }
    else if (r < 176) { int r2 = r - 48;  W = gW2 + (size_t)s * 256 * 256; nt = r2 / 8; kb = r2 % 8; fo = 256; }
    else              { int r3 = r - 176; W = gW3 + (size_t)s * 256 * 96;  nt = r3 / 8; kb = r3 % 8; fo = 96;  }
  } else if (b < FR_PL2) { int r = b - FR_PL1; W = pW1; nt = r / 3; kb = r % 3; fo = 256; }
  else                   { int r = b - FR_PL2; W = pW2; nt = r / 8; kb = r % 8; fo = 256; }

  int u    = threadIdx.x;         // 0..255
  int lane = u >> 3;              // 0..31
  int v    = u & 7;               // 0..7
  int half = lane >> 4, col = lane & 15;
  int k = kb * 32 + ((v >> 2) << 4) + (half << 3) + ((v & 3) << 1);
  int n = nt * 16 + col;
  _Float16 h0 = (_Float16)W[(size_t)k       * fo + n];
  _Float16 h1 = (_Float16)W[(size_t)(k + 1) * fo + n];
  size_t base = (size_t)b * 512;          // halfs
  ws[base + (size_t)lane * 16 + v * 2]     = h0;
  ws[base + (size_t)lane * 16 + v * 2 + 1] = h1;
}

// Branch-free tanh: no EXEC-mask branches (exp + rcp + copysign).
__device__ __forceinline__ float tanh_fast(float x) {
  float e = __expf(-2.0f * fabsf(x));              // in (0,1], never overflows
  float t = (1.0f - e) * __builtin_amdgcn_rcpf(1.0f + e);
  return copysignf(t, x);
}

// ---------------------------------------------------------------------------
// A-fragment from f16 LDS tile (ISA 7.12.2, 16-bit A 16x32):
// lane l -> M = l%16; VGPR v -> K = 16*(v/4) + 8*(l/16) + 2*(v%4) (+kb*32)
// Per lane the 16 halfs are two contiguous 16B runs -> exactly 2x ds_load_b128.
// ---------------------------------------------------------------------------
__device__ __forceinline__ v16h a_frag16(const _Float16* src, int stride, int lane, int kb) {
  int half = lane >> 4, m = lane & 15;
  const _Float16* row = src + m * stride + kb * 32;
  v8h lo = *(const v8h*)(row + (half << 3));
  v8h hi = *(const v8h*)(row + 16 + (half << 3));
  return __builtin_shufflevector(lo, hi, 0, 1, 2, 3, 4, 5, 6, 7,
                                 8, 9, 10, 11, 12, 13, 14, 15);
}

template <int KB>
__device__ __forceinline__ v8f mm_a16(const _Float16* A, int astride,
                                      const _Float16* wb, int lane) {
  v8f c = {};
#pragma unroll
  for (int kb = 0; kb < KB; ++kb) {
    v16h a = a_frag16(A, astride, lane, kb);
    v16h b = ((const v16h*)(wb + (size_t)kb * 512))[lane];
    c = __builtin_amdgcn_wmma_f32_16x16x32_f16(false, a, false, b, (short)0, c, false, false);
  }
  return c;
}

// C/D layout stores (lane l -> N = l%16, M = v + 8*(l/16)).
__device__ __forceinline__ void store_tanh_h(_Float16* dst, int dstride, int lane,
                                             int nt, const float* bias, v8f c) {
  int n  = nt * 16 + (lane & 15);
  int mb = (lane >> 4) * 8;
  float bv = bias[n];
#pragma unroll
  for (int v = 0; v < 8; ++v)
    dst[(mb + v) * dstride + n] = (_Float16)tanh_fast(c[v] + bv);
}

__device__ __forceinline__ void store_f32(float* dst, int dstride, int lane,
                                          int nt, const float* bias, v8f c) {
  int n  = nt * 16 + (lane & 15);
  int mb = (lane >> 4) * 8;
  float bv = bias[n];
#pragma unroll
  for (int v = 0; v < 8; ++v)
    dst[(mb + v) * dstride + n] = c[v] + bv;
}

// ---------------------------------------------------------------------------
// Persistent SDE kernel: 1 block = 16 batch rows, full 100-step loop.
// ---------------------------------------------------------------------------
__global__ void __launch_bounds__(256)
sde_kernel(const float* __restrict__ x0,  const float* __restrict__ dBt,
           const float* __restrict__ Dd,  const float* __restrict__ pb1,
           const float* __restrict__ pb2, const float* __restrict__ pW3,
           const float* __restrict__ pb3, const float* __restrict__ gb1,
           const float* __restrict__ gb2, const float* __restrict__ gb3,
           const _Float16* __restrict__ ws, float* __restrict__ out) {
  __shared__ float    sX [MT * XSTR];                 // xt tile (f32, exact state)
  __shared__ alignas(16) _Float16 sXh[MT * XHSTR];    // xt mirror (f16, WMMA A)
  __shared__ alignas(16) _Float16 sH [MT * HSTRH];    // hidden 1 (f16)
  __shared__ alignas(16) _Float16 sH2[MT * HSTRH];    // hidden 2 (f16)
  __shared__ float    sG [MT * XSTR];                 // grad_p tile (f32)
  __shared__ float sP[MT], sRel[MT], sRed[256];
  __shared__ int   sRun[MT], sHit[MT];

  const int t    = threadIdx.x;
  const int lane = t & 31;
  const int w    = t >> 5;            // wave 0..7
  const int blk  = blockIdx.x;
  const float dt    = 1.0f / NSTEP;
  const float sqdt  = sqrtf(dt);
  const float CLIPV = 1.0f - 1e-6f;

  // load x0 tile (rows are contiguous: 16*96 floats), fill f32 + f16 mirror
  for (int i = t; i < MT * D3DIM; i += 256) {
    float v = x0[(size_t)blk * MT * D3DIM + i];
    int r = i / D3DIM, c = i % D3DIM;
    sX [r * XSTR  + c] = v;
    sXh[r * XHSTR + c] = (_Float16)v;
  }
  __syncthreads();

  // ---- p0 = MLP(x0) with price-net weights ----
  {
    const _Float16* b1 = ws + (size_t)FR_PL1 * 512;
#pragma unroll
    for (int rep = 0; rep < 2; ++rep) {
      int nt = w + rep * 8;
      v8f c = mm_a16<3>(sXh, XHSTR, b1 + (size_t)(nt * 3) * 512, lane);
      store_tanh_h(sH, HSTRH, lane, nt, pb1, c);
    }
    __syncthreads();
    const _Float16* b2 = ws + (size_t)FR_PL2 * 512;
#pragma unroll
    for (int rep = 0; rep < 2; ++rep) {
      int nt = w + rep * 8;
      v8f c = mm_a16<8>(sH, HSTRH, b2 + (size_t)(nt * 8) * 512, lane);
      store_tanh_h(sH2, HSTRH, lane, nt, pb2, c);
    }
    __syncthreads();
    if (t < MT) {
      float acc = pb3[0];
      for (int k = 0; k < HID; ++k) acc += (float)sH2[t * HSTRH + k] * pW3[k];
      sP[t] = acc; sRun[t] = 1; sRel[t] = 0.0f;
    }
  }

  for (int s = 0; s < NSTEP; ++s) {
    __syncthreads();
    if (t < MT) sHit[t] = 0;
    const _Float16* sb = ws + (size_t)s * FR_PER_STEP * 512;

    // ---- grad_p = MLP(xt), per-step weights (all layers cvt-free f16 A) ----
#pragma unroll
    for (int rep = 0; rep < 2; ++rep) {
      int nt = w + rep * 8;
      v8f c = mm_a16<3>(sXh, XHSTR, sb + (size_t)(nt * 3) * 512, lane);
      store_tanh_h(sH, HSTRH, lane, nt, gb1 + s * HID, c);
    }
    __syncthreads();
#pragma unroll
    for (int rep = 0; rep < 2; ++rep) {
      int nt = w + rep * 8;
      v8f c = mm_a16<8>(sH, HSTRH, sb + (size_t)(48 + nt * 8) * 512, lane);
      store_tanh_h(sH2, HSTRH, lane, nt, gb2 + s * HID, c);
    }
    __syncthreads();
    if (w < 6) {
      v8f c = mm_a16<8>(sH2, HSTRH, sb + (size_t)(176 + w * 8) * 512, lane);
      store_f32(sG, XSTR, lane, w, gb3 + s * D3DIM, c);
    }
    __syncthreads();

    // ---- particle phase: 2 particles per thread; identity-based trig ----
    {
      int r = t >> 4, sub = t & 15;
      int rowg = blk * MT + r;
      float dp_part = 0.0f;
      int hit = 0;
      float runf = sRun[r] ? 1.0f : 0.0f;
#pragma unroll
      for (int j = 0; j < 2; ++j) {
        int q = sub * 2 + j;
        float* xp = &sX[r * XSTR + q * 3];
        float xx = xp[0], yy = xp[1], zz = xp[2];
        size_t di = (((size_t)s * BTOT + rowg) * DXP + q) * 2;
        float s2d = sqrtf(2.0f * Dd[q]) * sqdt;
        float dp0 = s2d * dBt[di];
        float dp1 = s2d * dBt[di + 1];
        float zc  = fminf(fmaxf(zz, -CLIPV), CLIPV);
        // ca = cos(acos(zc)-pi/2) = sqrt(1-zc^2);  sa = -zc
        float ca = sqrtf(fmaxf(1.0f - zc * zc, 0.0f));
        float sa = -zc;
        // cp = cos(atan2(y,x)) = x/r2;  sp = y/r2
        float rr = fmaxf(xx * xx + yy * yy, 1e-30f);
        float rn = __builtin_amdgcn_rsqf(rr);
        float cp = xx * rn, sp = yy * rn;
        // th = dp0+pi/2: sin(th)=cos(dp0), cos(th)=-sin(dp0)
        float cd0 = __cosf(dp0), sd0 = __sinf(dp0);
        float cd1 = __cosf(dp1), sd1 = __sinf(dp1);
        float st = cd0;
        float e0 = st * cd1 - 1.0f;
        float e1 = st * sd1;
        float e2 = -sd0;
        float d0 =  cp * ca * e0 - sp * e1 + cp * sa * e2;
        float d1 =  sp * ca * e0 + cp * e1 + sp * sa * e2;
        float d2 = -sa * e0 + ca * e2;
        xx += runf * d0; yy += runf * d1; zz += runf * d2;
        zz = fabsf(zz);
        xp[0] = xx; xp[1] = yy; xp[2] = zz;
        _Float16* xh = &sXh[r * XHSTR + q * 3];
        xh[0] = (_Float16)xx; xh[1] = (_Float16)yy; xh[2] = (_Float16)zz;
        if (zz < 0.05f) hit = 1;
        float g0 = sG[r * XSTR + q * 3], g1 = sG[r * XSTR + q * 3 + 1], g2 = sG[r * XSTR + q * 3 + 2];
        float r1 = -g0 * sp + g1 * cp;
        float r2 =  g0 * cp * sa + g1 * sp * sa + g2 * ca;
        dp_part += (-r2) * dp0 + r1 * dp1;
      }
      sRed[t] = dp_part;
      if (hit) atomicOr(&sHit[r], 1);
    }
    __syncthreads();

    // ---- row leaders: p update, hit handling (deterministic sum) ----
    if (t < MT && sRun[t]) {
      float sum = 0.0f;
      for (int i = 0; i < 16; ++i) sum += sRed[t * 16 + i];
      float dp = -0.2f * sP[t] * dt + sum;
      sP[t] += dp;
      if (sHit[t]) {
        float m = 0.0f;
        for (int i = 0; i < D3DIM; ++i) m += sX[t * XSTR + i];
        sRel[t] = m / (float)D3DIM;
        sRun[t] = 0;
      }
    }
  }
  __syncthreads();

  if (t < MT) {
    if (sRun[t]) {
      float m = 0.0f;
      for (int i = 0; i < D3DIM; ++i) m += sX[t * XSTR + i];
      sRel[t] = m / (float)D3DIM;
    }
    size_t rowg = (size_t)blk * MT + t;
    out[rowg * 2]     = sP[t];
    out[rowg * 2 + 1] = sRel[t];
  }
}

extern "C" void kernel_launch(void* const* d_in, const int* in_sizes, int n_in,
                              void* d_out, int out_size, void* d_ws, size_t ws_size,
                              hipStream_t stream) {
  const float* x0  = (const float*)d_in[0];
  const float* dBt = (const float*)d_in[1];
  const float* Dd  = (const float*)d_in[2];
  const float* pW1 = (const float*)d_in[3];
  const float* pb1 = (const float*)d_in[4];
  const float* pW2 = (const float*)d_in[5];
  const float* pb2 = (const float*)d_in[6];
  const float* pW3 = (const float*)d_in[7];
  const float* pb3 = (const float*)d_in[8];
  const float* gW1 = (const float*)d_in[9];
  const float* gb1 = (const float*)d_in[10];
  const float* gW2 = (const float*)d_in[11];
  const float* gb2 = (const float*)d_in[12];
  const float* gW3 = (const float*)d_in[13];
  const float* gb3 = (const float*)d_in[14];
  _Float16* ws = (_Float16*)d_ws;

  pack_weights<<<FR_TOTAL, 256, 0, stream>>>(gW1, gW2, gW3, pW1, pW2, ws);
  sde_kernel<<<BTOT / MT, 256, 0, stream>>>(x0, dBt, Dd, pb1, pb2, pW3, pb3,
                                            gb1, gb2, gb3, ws, (float*)d_out);
}
